// SentimentModel_46651934769817
// MI455X (gfx1250) — matrix operand order repaired and checked by reading, not compiled, not run on hardware
//
#include <hip/hip_runtime.h>

#define B_  64
#define S_  512
#define E_  512
#define H_  1024
#define G4_ 4096   // 4*H
#define NBLK_ 32   // persistent grid size

typedef __attribute__((ext_vector_type(16))) __bf16          v16bf;
typedef __attribute__((ext_vector_type(8)))  float           v8f;
typedef __attribute__((ext_vector_type(16))) unsigned short  v16u;
typedef __attribute__((ext_vector_type(8)))  unsigned short  v8u;

union BF16x16 {
    v16bf bf;
    v16u  u16;
    unsigned short u[16];
};

__device__ __forceinline__ unsigned short f32_to_bf16(float f) {
    union { float f; unsigned int u; } x; x.f = f;
    unsigned int r = x.u + 0x7FFFu + ((x.u >> 16) & 1u);   // round-to-nearest-even
    return (unsigned short)(r >> 16);
}

__device__ __forceinline__ float fast_sigmoid(float x) {
    return 1.0f / (1.0f + __expf(-x));
}
__device__ __forceinline__ float fast_tanh(float x) {
    x = fminf(15.0f, fmaxf(-15.0f, x));
    float e = __expf(2.0f * x);
    return (e - 1.0f) / (e + 1.0f);
}

// ---------------------------------------------------------------- init kernels
__global__ void convert_f32_bf16(const float* __restrict__ in,
                                 unsigned short* __restrict__ out, int n) {
    int i = blockIdx.x * blockDim.x + threadIdx.x;
    if (i < n) out[i] = f32_to_bf16(in[i]);
}

__global__ void zero_u32(unsigned int* p) { *p = 0u; }

// One-time: materialize emb[x] as bf16 [B,S,E] so the recurrent scan is pure bf16.
__global__ void gather_embed_bf16(const int* __restrict__ x,        // [B*S]
                                  const float* __restrict__ emb,    // [V,E]
                                  unsigned short* __restrict__ out, // [B*S,E]
                                  int n) {                          // n = B*S*E
    int i = blockIdx.x * blockDim.x + threadIdx.x;
    if (i < n) {
        int row = i >> 9;            // / E_
        int col = i & (E_ - 1);      // % E_
        int token = x[row];
        out[i] = f32_to_bf16(emb[(size_t)token * E_ + col]);
    }
}

// ---------------------------------------------------------------- grid barrier
// Monotonic phase barrier: each use adds gridDim.x to the counter; a block waits
// until the counter reaches the next multiple. Only thread 0 spins (s_sleep).
__device__ __forceinline__ void grid_barrier(unsigned int* cnt, unsigned int nblocks) {
    __syncthreads();
    if (threadIdx.x == 0) {
        __threadfence();  // release prior global writes (agent scope)
        unsigned int my = __hip_atomic_fetch_add(cnt, 1u, __ATOMIC_RELEASE,
                                                 __HIP_MEMORY_SCOPE_AGENT);
        unsigned int target = (my / nblocks + 1u) * nblocks;
        while (__hip_atomic_load(cnt, __ATOMIC_ACQUIRE,
                                 __HIP_MEMORY_SCOPE_AGENT) < target) {
            __builtin_amdgcn_s_sleep(8);
        }
    }
    __syncthreads();
}

// ---------------------------------------------------------------- GEMM phase helper
template<int K, int LDW>
__device__ __forceinline__ void gemm_phase(const unsigned short* __restrict__ arow,
                                           const unsigned short* __restrict__ W,
                                           const size_t grow[4],
                                           int hi, v8f acc[4]) {
    #pragma unroll 2
    for (int kb = 0; kb < K; kb += 32) {
        BF16x16 A;
        A.u16 = *(const v16u*)(arow + kb + hi * 16);
        #pragma unroll
        for (int q = 0; q < 4; ++q) {
            BF16x16 Bv;
            Bv.u16 = *(const v16u*)(W + grow[q] * LDW + kb + hi * 16);
            acc[q] = __builtin_amdgcn_wmma_f32_16x16x32_bf16(
                false, A.bf, false, Bv.bf, (short)0, acc[q], false, false);
        }
    }
}

// ---------------------------------------------------------------- persistent LSTM scan
// 256 waves: wave = m0 (batch tile 0..3) * 64 + jt (h-col tile 0..63).
// One kernel runs all 512 timesteps; cell state c lives in registers for the
// entire scan; steps are separated by a device-wide barrier; h is double-buffered
// in bf16 through L2.
__global__ __launch_bounds__(256)
void lstm_scan(const unsigned short* __restrict__ embbf,    // [B,S,E] bf16
               const unsigned short* __restrict__ Wih,      // [4H,E] bf16
               const unsigned short* __restrict__ Whh,      // [4H,H] bf16
               const float*          __restrict__ b_ih,     // [4H]
               const float*          __restrict__ b_hh,     // [4H]
               const float*          __restrict__ c0,       // [B,H] f32
               unsigned short*       __restrict__ hbf0,     // [B,H] bf16 (init = h0)
               unsigned short*       __restrict__ hbf1,     // [B,H] bf16
               float*                __restrict__ c_out,    // [B,H] f32 (d_out)
               float*                __restrict__ h_out_f32,// [B,H] f32 (d_out)
               unsigned int*         __restrict__ bar)
{
    const int lane = threadIdx.x & 31;
    const int wave = blockIdx.x * (blockDim.x >> 5) + (threadIdx.x >> 5);
    const int m0   = wave >> 6;          // batch tile 0..3
    const int hc0  = (wave & 63) << 4;   // h-column base

    const int n  = lane & 15;
    const int hi = lane >> 4;

    const int mrow = m0 * 16 + n;        // A row for this lane (both halves)
    const unsigned short* erow = embbf + (size_t)mrow * S_ * E_;  // advance by E_ per step

    size_t grow[4];
    #pragma unroll
    for (int q = 0; q < 4; ++q) grow[q] = (size_t)(q * H_ + hc0 + n);

    float bias[4];
    #pragma unroll
    for (int q = 0; q < 4; ++q) bias[q] = b_ih[grow[q]] + b_hh[grow[q]];

    // this lane's 8 cell/hidden elements: (m = m0*16 + r + 8*hi, hc = hc0 + n)
    const int hc = hc0 + n;
    size_t idx[8];
    float  creg[8], hreg[8];
    #pragma unroll
    for (int r = 0; r < 8; ++r) {
        idx[r]  = (size_t)(m0 * 16 + r + 8 * hi) * H_ + hc;
        creg[r] = c0[idx[r]];
        hreg[r] = 0.0f;
    }

    for (int s = 0; s < S_; ++s) {
        const unsigned short* hin  = (s & 1) ? hbf1 : hbf0;
        unsigned short*       hout = (s & 1) ? hbf0 : hbf1;
        const unsigned short* hrow = hin + (size_t)mrow * H_;

        v8f acc[4];
        #pragma unroll
        for (int q = 0; q < 4; ++q) acc[q] = {};

        // gates  = embed(x_s) @ W_ih^T   (K = E)
        gemm_phase<E_, E_>(erow, Wih, grow, hi, acc);
        // gates += h        @ W_hh^T   (K = H)
        gemm_phase<H_, H_>(hrow, Whh, grow, hi, acc);

        #pragma unroll
        for (int r = 0; r < 8; ++r) {
            float iv = fast_sigmoid(acc[0][r] + bias[0]);
            float fv = fast_sigmoid(acc[1][r] + bias[1]);
            float gv = fast_tanh   (acc[2][r] + bias[2]);
            float ov = fast_sigmoid(acc[3][r] + bias[3]);
            float c2 = fv * creg[r] + iv * gv;
            float h2 = ov * fast_tanh(c2);
            creg[r] = c2;
            hreg[r] = h2;
            hout[idx[r]] = f32_to_bf16(h2);
        }

        erow += E_;
        grid_barrier(bar, NBLK_);   // writes of hout visible before next step reads
    }

    // final state to d_out
    #pragma unroll
    for (int r = 0; r < 8; ++r) {
        h_out_f32[idx[r]] = hreg[r];
        c_out[idx[r]]     = creg[r];
    }
}

// ---------------------------------------------------------------- final FC + sigmoid
__global__ void final_fc(const float* __restrict__ h,
                         const float* __restrict__ fc_w,
                         const float* __restrict__ fc_b,
                         float* __restrict__ sig_out)
{
    int b = threadIdx.x;
    if (b < B_) {
        float acc = fc_b[0];
        for (int k = 0; k < H_; ++k) acc += h[(size_t)b * H_ + k] * fc_w[k];
        sig_out[b] = fast_sigmoid(acc);
    }
}

// ---------------------------------------------------------------- launch
extern "C" void kernel_launch(void* const* d_in, const int* in_sizes, int n_in,
                              void* d_out, int out_size, void* d_ws, size_t ws_size,
                              hipStream_t stream) {
    const int*   x     = (const int*)  d_in[0];
    const float* h0    = (const float*)d_in[1];
    const float* c0    = (const float*)d_in[2];
    const float* emb   = (const float*)d_in[3];
    const float* Wih_f = (const float*)d_in[4];
    const float* Whh_f = (const float*)d_in[5];
    const float* b_ih  = (const float*)d_in[6];
    const float* b_hh  = (const float*)d_in[7];
    const float* fc_w  = (const float*)d_in[8];
    const float* fc_b  = (const float*)d_in[9];

    float* out   = (float*)d_out;
    float* sig   = out;                     // [64]
    float* h_f32 = out + 64;                // [B,H]
    float* c_out = out + 64 + B_ * H_;      // [B,H]

    unsigned short* Wih   = (unsigned short*)d_ws;            // 4096*512   (4 MB)
    unsigned short* Whh   = Wih   + (size_t)G4_ * E_;         // 4096*1024  (8 MB)
    unsigned short* embbf = Whh   + (size_t)G4_ * H_;         // 64*512*512 (32 MB)
    unsigned short* hbf0  = embbf + (size_t)B_ * S_ * E_;     // 64*1024
    unsigned short* hbf1  = hbf0  + (size_t)B_ * H_;          // 64*1024
    unsigned int*   bar   = (unsigned int*)(hbf1 + (size_t)B_ * H_);  // barrier counter

    // per-call re-init (deterministic for graph replay)
    convert_f32_bf16<<<(G4_ * E_ + 255) / 256, 256, 0, stream>>>(Wih_f, Wih, G4_ * E_);
    convert_f32_bf16<<<(G4_ * H_ + 255) / 256, 256, 0, stream>>>(Whh_f, Whh, G4_ * H_);
    convert_f32_bf16<<<(B_  * H_ + 255) / 256, 256, 0, stream>>>(h0, hbf0, B_ * H_);
    zero_u32<<<1, 1, 0, stream>>>(bar);
    {
        const int n = B_ * S_ * E_;
        gather_embed_bf16<<<(n + 255) / 256, 256, 0, stream>>>(x, emb, embbf, n);
    }

    // persistent recurrent scan: one launch for all 512 steps
    lstm_scan<<<NBLK_, 256, 0, stream>>>(embbf, Wih, Whh, b_ih, b_hh, c0,
                                         hbf0, hbf1, c_out, h_f32, bar);

    final_fc<<<1, 64, 0, stream>>>(h_f32, fc_w, fc_b, sig);
}
